// FusionLoss_30227979829940
// MI455X (gfx1250) — compile-verified, matrix operand order checked by reading, not run on hardware
//
#include <hip/hip_runtime.h>
#include <hip/hip_bf16.h>

// Fused FusionLoss for MI455X (gfx1250).
// Latency-bound problem (~0.3 MB of real HBM traffic at 23.3 TB/s; launch latency
// dominates): one kernel, one workgroup, all intermediates in LDS, deterministic
// tree reduction (no float atomics).
// The reference's only matmul (l@G and v@G group sums, (128x12)@(12x4)) is mapped
// onto V_WMMA_F32_16X16X4_F32 (exact f32): 16 batch rows per wave, K=12 split into
// three K=4 chunks, two f32 accumulators (sum_l, num) -> 6 WMMAs per wave tile.
// Edge tables are bit-packed immediates (shift/and/cndmask lookups, no memory).
// num is an exact integer in {0..4}, so 1/max(num,1) is a 4-way cndmask select.
// sqrt uses the raw V_SQRT_F32 (~1 ulp) instead of the refined libm expansion.

#define BDIM 256
constexpr int Bn = 128, Jn = 16, Hn = 128, Wn = 128;
constexpr int HW = Hn * Wn;
constexpr int NE = 12, NG = 4;

// Packed per-edge tables (edge e in nibble/2-bit slot e, LSB first):
// ID1 = {0,1,3,4,10,11,13,14,2,3,12,13}
constexpr unsigned long long PID1 = 0xDC32EDBA4310ULL;
// ID2 = {1,2,4,5,11,12,14,15,6,6,8,8}
constexpr unsigned long long PID2 = 0x8866FECB5421ULL;
// GID = {0,0,0,0,1,1,1,1,2,2,3,3}  (2 bits per edge)
constexpr unsigned int PGID = 0x00FA5500u;
// WSK code: 0 -> 1.0, 1 -> 1.0085885..., 2 -> 1.1375361...  (2 bits per edge)
constexpr unsigned int PWSK = 0x00008241u;
constexpr float WSK1 = 1.0085885098415446f;
constexpr float WSK2 = 1.1375361376887123f;

typedef float    v2f  __attribute__((ext_vector_type(2)));
typedef float    v8f  __attribute__((ext_vector_type(8)));
typedef _Float16 v16h __attribute__((ext_vector_type(16)));

__global__ __launch_bounds__(BDIM) void fusion_loss_kernel(
    const float* __restrict__ output,   // (B,J,H,W) f32
    const int*   __restrict__ mask,     // (B,J) i32
    const int*   __restrict__ ind,      // (B,J) i32, in [0, J*H*W)
    const float* __restrict__ target,   // (B,J,1) f32   (also "visible")
    const float* __restrict__ gt2d,     // (B,2J) f32
    float*       __restrict__ out)      // scalar
{
    __shared__ float s_pred[Bn][Jn];    // gathered predictions
    __shared__ float s_l[Bn][NE];       // per-edge l
    __shared__ float s_v[Bn][NE];       // per-edge visibility (0/1)
    __shared__ float s_E[Bn][NG];       // per-group mean
    __shared__ float s_rn[Bn][NG];      // per-group 0.5 / max(num,1)
    __shared__ float s_active[Bn];      // row active flag
    __shared__ float s_r0[BDIM], s_r1[BDIM], s_r2[BDIM];

    const int tid  = threadIdx.x;
    const int lane = tid & 31;
    const int wid  = tid >> 5;          // 8 waves; wave w owns batch rows 16w..16w+15

    // ---------------- Phase 1: gather + reg-loss partials --------------------
    float sl1 = 0.0f, nmask = 0.0f;
    for (int i = tid; i < Bn * Jn; i += BDIM) {       // exactly 8 iters/thread
        const int b  = i >> 4;
        const int j  = i & 15;
        const int ix = ind[i];
        const int jj = ix & 15;                        // ix % J   (J==16)
        const int hw = ix >> 4;                        // ix / J -> h*W + w
        const float p = output[(size_t)b * (Jn * HW) + (size_t)jj * HW + hw];
        s_pred[b][j] = p;
        const float m  = (float)mask[i];
        nmask += m;
        const float d  = p * m - target[i] * m;
        const float ad = fabsf(d);
        sl1 += (ad < 1.0f) ? 0.5f * d * d : (ad - 0.5f);
    }
    __syncthreads();

    if (tid < Bn) {
        int ms = 0;
        for (int j = 0; j < Jn; ++j) ms += mask[tid * Jn + j];
        s_active[tid] = (ms == 0) ? 1.0f : 0.0f;
    }

    // ---------------- Phase 2: edge terms + WMMA group sums ------------------
    const int base = wid * 16;
    const int row  = base + (lane & 15);
    const int col  = lane & 15;                        // B-matrix column = group id

    // Branchless per-edge l, v (all table lookups from packed immediates).
    auto edge_lv = [&](int r, int e, float& lv, float& vv) {
        const int i1 = (int)((PID1 >> (4 * e)) & 0xF);
        const int i2 = (int)((PID2 >> (4 * e)) & 0xF);
        const float vis1 = target[r * Jn + i1];
        const float vis2 = target[r * Jn + i2];
        const float dx = gt2d[r * 2 * Jn + 2 * i1]     - gt2d[r * 2 * Jn + 2 * i2];
        const float dy = gt2d[r * 2 * Jn + 2 * i1 + 1] - gt2d[r * 2 * Jn + 2 * i2 + 1];
        const float dz = s_pred[r][i1] - s_pred[r][i2];
        const bool  vb = (vis1 > 0.5f) && (vis2 > 0.5f);
        const float ss = dx * dx + dy * dy + dz * dz;
        const float s  = vb ? ss : 1.0f;               // cndmask, no branch
        const int   wc = (int)((PWSK >> (2 * e)) & 3);
        const float w  = (wc == 1) ? WSK1 : ((wc == 2) ? WSK2 : 1.0f);
        vv = vb ? 1.0f : 0.0f;
        // Raw V_SQRT_F32 (~1 ulp); argument is >= 0 by construction, sqrt(1)=1
        // in the invisible case so the result is always well-defined.
        lv = vv * (__builtin_amdgcn_sqrtf(s) * w);
    };

    v8f acc_l = {};   // sum_l per (row, group)
    v8f acc_v = {};   // num   per (row, group)

#if __has_builtin(__builtin_amdgcn_wmma_f32_16x16x4_f32)
    // Exact-f32 path: 3 chunks of K=4 (edges 4c..4c+3).
    // A 16x4 f32 layout: lanes 0-15 -> K=0,1 ; lanes 16-31 -> K=2,3. B mirrors it.
    const int koff = (lane < 16) ? 0 : 2;
#pragma unroll
    for (int c = 0; c < 3; ++c) {
        const int e0 = 4 * c + koff, e1 = e0 + 1;
        float l0, v0, l1, v1;
        edge_lv(row, e0, l0, v0);
        edge_lv(row, e1, l1, v1);
        s_l[row][e0] = l0; s_v[row][e0] = v0;
        s_l[row][e1] = l1; s_v[row][e1] = v1;
        const int g0 = (int)((PGID >> (2 * e0)) & 3);
        const int g1 = (int)((PGID >> (2 * e1)) & 3);
        const float b0 = (col == g0) ? 1.0f : 0.0f;    // col==g implies col<4
        const float b1 = (col == g1) ? 1.0f : 0.0f;
        v2f Al = {l0, l1}, Av = {v0, v1}, Bm = {b0, b1};
        acc_l = __builtin_amdgcn_wmma_f32_16x16x4_f32(false, Al, false, Bm,
                                                      (short)0, acc_l, false, false);
        acc_v = __builtin_amdgcn_wmma_f32_16x16x4_f32(false, Av, false, Bm,
                                                      (short)0, acc_v, false, false);
    }
#else
    // Fallback: codegen-confirmed f16 WMMA, K=12 real values zero-padded to 32.
    v16h Al = {}, Av = {}, Bf = {};
    if (lane < 16) {
        for (int e = 0; e < 8; ++e) {
            float lv, vv; edge_lv(row, e, lv, vv);
            s_l[row][e] = lv; s_v[row][e] = vv;
            Al[e] = (_Float16)lv;
            Av[e] = (_Float16)vv;
            const int g = (int)((PGID >> (2 * e)) & 3);
            Bf[e] = (_Float16)((col == g) ? 1.0f : 0.0f);
        }
    } else {
        for (int e = 0; e < 4; ++e) {
            const int ee = 8 + e;
            float lv, vv; edge_lv(row, ee, lv, vv);
            s_l[row][ee] = lv; s_v[row][ee] = vv;
            Al[e] = (_Float16)lv;
            Av[e] = (_Float16)vv;
            const int g = (int)((PGID >> (2 * ee)) & 3);
            Bf[e] = (_Float16)((col == g) ? 1.0f : 0.0f);
        }
    }
    acc_l = __builtin_amdgcn_wmma_f32_16x16x32_f16(false, Al, false, Bf,
                                                   (short)0, acc_l, false, false);
    acc_v = __builtin_amdgcn_wmma_f32_16x16x32_f16(false, Av, false, Bf,
                                                   (short)0, acc_v, false, false);
#endif

    // C/D 16x16 f32 layout: VGPR r, lanes 0-15 -> (M=r, N=lane); lanes 16-31 -> (M=8+r, N=lane-16)
    // num is an exact integer in {0..4} -> reciprocal via 4-way select (no divider).
    if (col < NG) {
        const int roff = (lane < 16) ? 0 : 8;
#pragma unroll
        for (int r = 0; r < 8; ++r) {
            const int rr = base + roff + r;
            const float nv = acc_v[r];
            const float sl = acc_l[r];
            const float rcp = (nv <= 1.5f) ? 1.0f
                            : (nv <= 2.5f) ? 0.5f
                            : (nv <= 3.5f) ? (1.0f / 3.0f) : 0.25f;
            s_E[rr][col]  = (nv >= 0.5f) ? (sl * rcp) : 0.0f;
            s_rn[rr][col] = 0.5f * rcp;                 // = 1 / (2*max(num,1))
        }
    }
    __syncthreads();

    // ---------------- Phase 3: variance terms (branchless, FMA only) ---------
    float varp = 0.0f;
    for (int i = tid; i < Bn * NE; i += BDIM) {        // exactly 6 iters/thread
        const int r = i / NE;
        const int e = i - r * NE;
        const int g = (int)((PGID >> (2 * e)) & 3);
        const float diff = s_l[r][e] - s_E[r][g];
        // v==0 -> contributes 0; s_rn is always finite.
        varp += s_v[r][e] * s_active[r] * (diff * diff) * s_rn[r][g];
    }

    // ---------------- Phase 4: deterministic block reduction -----------------
    s_r0[tid] = sl1;  s_r1[tid] = nmask;  s_r2[tid] = varp;
    __syncthreads();
    for (int s = BDIM / 2; s > 0; s >>= 1) {
        if (tid < s) {
            s_r0[tid] += s_r0[tid + s];
            s_r1[tid] += s_r1[tid + s];
            s_r2[tid] += s_r2[tid + s];
        }
        __syncthreads();
    }
    if (tid == 0) {
        const float reg = s_r0[0] / (s_r1[0] + 0.0001f);     // REG_WEIGHT = 1.0
        const float var = 0.01f * s_r2[0] / (float)Bn;       // VAR_WEIGHT = 0.01
        out[0] = reg + var;
    }
}

extern "C" void kernel_launch(void* const* d_in, const int* in_sizes, int n_in,
                              void* d_out, int out_size, void* d_ws, size_t ws_size,
                              hipStream_t stream) {
    (void)in_sizes; (void)n_in; (void)out_size; (void)d_ws; (void)ws_size;
    const float* output = (const float*)d_in[0];
    const int*   mask   = (const int*)  d_in[1];
    const int*   ind    = (const int*)  d_in[2];
    const float* target = (const float*)d_in[3];
    const float* gt2d   = (const float*)d_in[4];
    fusion_loss_kernel<<<dim3(1), dim3(BDIM), 0, stream>>>(
        output, mask, ind, target, gt2d, (float*)d_out);
}
